// Pointnet2_Large_9672266350687
// MI455X (gfx1250) — compile-verified, matrix operand order checked by reading, not run on hardware
//
#include <hip/hip_runtime.h>
#include <math.h>

#define STR2(x) #x
#define STR(x) STR2(x)
#pragma message("CDNA5 toolchain probe: clang " STR(__clang_major__) "." STR(__clang_minor__))

typedef __bf16 bf16;
typedef __attribute__((ext_vector_type(16))) __bf16 v16bf;
typedef __attribute__((ext_vector_type(8)))  __bf16 v8bf;
typedef __attribute__((ext_vector_type(8)))  float  v8f;
typedef const __bf16 __attribute__((address_space(3))) lds_cbf16;

#define BATCH 2
#define NPTS  8192
#define NCLS  13
#define BN_SCALE_F 0.9999950000374997f

static inline int align32i(int c) { return (c + 31) & ~31; }

// ---------------------------------------------------------------------------
// CDNA5 async global->LDS staging (ASYNCcnt-tracked, writes LDS directly,
// no VGPR round trip).  LDS operand is the 32-bit LDS byte offset
// (generic->AS3 cast folds to the raw offset).
// ---------------------------------------------------------------------------
__device__ __forceinline__ void async_b128(const bf16* g, const bf16* l) {
    unsigned loff = (unsigned)(unsigned long long)(lds_cbf16*)l;
    asm volatile("global_load_async_to_lds_b128 %0, %1, off"
                 :: "v"(loff), "v"((unsigned long long)g)
                 : "memory");
}
#define S_WAIT_ASYNCCNT(n) asm volatile("s_wait_asynccnt " #n ::: "memory")

// ---------------------------------------------------------------------------
// Normalization: per-(b, coord) mean / std (ddof=1) over N points, then
// x = (xyz - mean) / std.  Input layout [B,3,N]; outputs point-major
// [B,N,3] f32 (for geometry) and bf16 (as l0 point features).
// ---------------------------------------------------------------------------
__global__ void meanstd_kernel(const float* __restrict__ xyz, float* __restrict__ mstd) {
    int bc = blockIdx.x;                       // b*3 + c, 6 blocks
    const float* p = xyz + (size_t)bc * NPTS;
    __shared__ float ss[256], sq[256];
    int t = threadIdx.x;
    float s = 0.f, q = 0.f;
    for (int i = t; i < NPTS; i += 256) { float v = p[i]; s += v; q += v * v; }
    ss[t] = s; sq[t] = q;
    __syncthreads();
    for (int o = 128; o > 0; o >>= 1) {
        if (t < o) { ss[t] += ss[t + o]; sq[t] += sq[t + o]; }
        __syncthreads();
    }
    if (t == 0) {
        float mean = ss[0] / (float)NPTS;
        float var = (sq[0] - (float)NPTS * mean * mean) / (float)(NPTS - 1);
        var = var > 0.f ? var : 0.f;
        float sd = sqrtf(var);
        if (sd == 0.f) sd = 1e-8f;
        mstd[bc * 2] = mean;
        mstd[bc * 2 + 1] = sd;
    }
}

__global__ void norm_kernel(const float* __restrict__ xyz, const float* __restrict__ mstd,
                            float* __restrict__ xyzT, bf16* __restrict__ pts0, int total) {
    int t = blockIdx.x * blockDim.x + threadIdx.x;
    if (t >= total) return;
    int b = t / (NPTS * 3);
    int rem = t - b * NPTS * 3;
    int n = rem / 3, c = rem - n * 3;
    float m = mstd[(b * 3 + c) * 2];
    float sd = mstd[(b * 3 + c) * 2 + 1];
    float v = (xyz[((size_t)b * 3 + c) * NPTS + n] - m) / sd;
    xyzT[t] = v;
    pts0[t] = (bf16)v;
}

// ---------------------------------------------------------------------------
// Farthest point sampling: one block per batch.  Running min-distance array
// kept resident in LDS (<=32KB), block-wide argmax each iteration.
// ---------------------------------------------------------------------------
__global__ __launch_bounds__(512) void fps_kernel(const float* __restrict__ xyz, int Nin, int S,
                                                  int* __restrict__ idxOut, float* __restrict__ newXyz) {
    int b = blockIdx.x;
    const float* X = xyz + (size_t)b * Nin * 3;
    __shared__ float dist[NPTS];
    __shared__ float rv[512];
    __shared__ int   ri[512];
    __shared__ int   sFar;
    int t = threadIdx.x;
    for (int i = t; i < Nin; i += 512) dist[i] = 1e10f;
    if (t == 0) sFar = 0;
    __syncthreads();
    for (int s = 0; s < S; s++) {
        int far = sFar;
        float cx = X[far * 3], cy = X[far * 3 + 1], cz = X[far * 3 + 2];
        if (t == 0) {
            idxOut[b * S + s] = far;
            float* nx = newXyz + ((size_t)b * S + s) * 3;
            nx[0] = cx; nx[1] = cy; nx[2] = cz;
        }
        float bv = -1.f; int bi = 0;
        for (int i = t; i < Nin; i += 512) {
            float dx = X[i * 3] - cx, dy = X[i * 3 + 1] - cy, dz = X[i * 3 + 2] - cz;
            float d = dx * dx + dy * dy + dz * dz;
            float nd = fminf(dist[i], d);
            dist[i] = nd;
            if (nd > bv) { bv = nd; bi = i; }
        }
        rv[t] = bv; ri[t] = bi;
        __syncthreads();
        for (int o = 256; o > 0; o >>= 1) {
            if (t < o) {
                if (rv[t + o] > rv[t] || (rv[t + o] == rv[t] && ri[t + o] < ri[t])) {
                    rv[t] = rv[t + o]; ri[t] = ri[t + o];
                }
            }
            __syncthreads();
        }
        if (t == 0) sFar = ri[0];
        __syncthreads();
    }
}

// ---------------------------------------------------------------------------
// Ball query: per query point, first K indices with d^2 <= r^2, padded with
// the first hit (matches reference semantics).
// ---------------------------------------------------------------------------
__global__ void ballquery_kernel(const float* __restrict__ xyz, const float* __restrict__ newXyz,
                                 int Nin, int S, int K, float r2, int* __restrict__ gidx, int total) {
    int t = blockIdx.x * blockDim.x + threadIdx.x;
    if (t >= total) return;
    int b = t / S, s = t - b * S;
    const float* q = newXyz + ((size_t)b * S + s) * 3;
    const float* X = xyz + (size_t)b * Nin * 3;
    int* out = gidx + (size_t)t * K;
    float qx = q[0], qy = q[1], qz = q[2];
    int cnt = 0;
    for (int n = 0; n < Nin && cnt < K; n++) {
        float dx = X[n * 3] - qx, dy = X[n * 3 + 1] - qy, dz = X[n * 3 + 2] - qz;
        if (dx * dx + dy * dy + dz * dz <= r2) out[cnt++] = n;
    }
    int f = (cnt > 0) ? out[0] : 0;
    for (; cnt < K; cnt++) out[cnt] = f;
}

// ---------------------------------------------------------------------------
// Grouping: gp[b,s,k,:] = concat(point_feats[gidx], xyz[gidx]-new_xyz),
// bf16 output with channel padding to a multiple of 32 (zeros).
// ---------------------------------------------------------------------------
__global__ void group_kernel(const float* __restrict__ xyz, const float* __restrict__ newXyz,
                             const bf16* __restrict__ pts, int ldp, int Cp,
                             const int* __restrict__ gidx, int Nin, int S, int K,
                             bf16* __restrict__ gp, int lda, long total) {
    long t = (long)blockIdx.x * blockDim.x + threadIdx.x;
    if (t >= total) return;
    int c = (int)(t % lda);
    long bsk = t / lda;
    long sk = (long)S * K;
    int b = (int)(bsk / sk);
    long rem = bsk - (long)b * sk;
    int s = (int)(rem / K);
    int idx = gidx[bsk];
    float v = 0.f;
    if (c < Cp) {
        v = (float)pts[((size_t)b * Nin + idx) * ldp + c];
    } else if (c < Cp + 3) {
        int d = c - Cp;
        v = xyz[((size_t)b * Nin + idx) * 3 + d] - newXyz[((size_t)b * S + s) * 3 + d];
    }
    gp[t] = (bf16)v;
}

// ---------------------------------------------------------------------------
// Weight conversion f32 [O,C] -> bf16 [O, Ka] with zero K-padding.
// ---------------------------------------------------------------------------
__global__ void wconv_kernel(const float* __restrict__ W, int C, bf16* __restrict__ Wb,
                             int Ka, long total) {
    long t = (long)blockIdx.x * blockDim.x + threadIdx.x;
    if (t >= total) return;
    int k = (int)(t % Ka);
    long o = t / Ka;
    Wb[t] = (bf16)((k < C) ? W[(size_t)o * C + k] : 0.f);
}

// ---------------------------------------------------------------------------
// WMMA GEMM with fused bias + batchnorm-scale + ReLU epilogue.
//   Out[m, o] = relu((sum_k A[m,k] * W[o,k] + bias[o]) * g[o]*BN_SCALE + be[o])
// Block = 256 threads = 8 waves; block tile 128(M) x 32(N).  Wave w owns rows
// w*16..w*16+15 and BOTH 16-wide N-subtiles -> 2 WMMAs / K-step sharing one
// A fragment.  A/B tiles double-buffered in LDS and staged with
// global_load_async_to_lds_b128 (ASYNCcnt), so stage i+1 loads overlap
// stage i math.  LDS row stride 40 bf16 keeps 16B alignment for ds_load_b128.
// Fragment layout per ISA 7.12.2 (16-bit, wave32).
// ---------------------------------------------------------------------------
union bffrag { v16bf v; v8bf h[2]; };

#define A_TILE (128 * 40)
#define B_TILE (32 * 40)

__global__ __launch_bounds__(256) void wmma_gemm_bnrelu(
        const bf16* __restrict__ A, const bf16* __restrict__ Wb,
        const float* __restrict__ bias, const float* __restrict__ g,
        const float* __restrict__ be,
        bf16* __restrict__ Out, int ldo, int Ka) {
    __shared__ __attribute__((aligned(16))) bf16 As[2 * A_TILE];
    __shared__ __attribute__((aligned(16))) bf16 Bs[2 * B_TILE];

    const int m0 = blockIdx.x * 128;
    const int n0 = blockIdx.y * 32;
    const int t = threadIdx.x;
    const int lane = t & 31;
    const int w = t >> 5;
    const int r = lane & 15;
    const int koff = (lane >> 4) * 8;   // lanes 0-15: K{0..7,16..23}; 16-31: K{8..15,24..31}

    // staging decomposition (per stage: 2 A-chunks/thread, 1 B-chunk for t<128)
    const int ar0 = t >> 2, ac0 = (t & 3) * 8;
    const int ar1 = (t + 256) >> 2, ac1 = ac0;
    const int br = t >> 2, bc = (t & 3) * 8;
    const bool doB = (t < 128);          // wave-uniform (waves 0-3)

    const int nStages = Ka >> 5;
    v8f acc0 = {}, acc1 = {};

    // issue stage 0
    {
        const bf16* Ag = A + (size_t)m0 * Ka;
        async_b128(Ag + (size_t)ar0 * Ka + ac0, As + ar0 * 40 + ac0);
        async_b128(Ag + (size_t)ar1 * Ka + ac1, As + ar1 * 40 + ac1);
        if (doB) async_b128(Wb + (size_t)(n0 + br) * Ka + bc, Bs + br * 40 + bc);
    }

    for (int s = 0; s < nStages; s++) {
        const int buf = s & 1;
        const bool hasNext = (s + 1) < nStages;
        if (hasNext) {
            const int nb = buf ^ 1;
            const int kb = (s + 1) << 5;
            const bf16* Ag = A + (size_t)m0 * Ka + kb;
            async_b128(Ag + (size_t)ar0 * Ka + ac0, As + nb * A_TILE + ar0 * 40 + ac0);
            async_b128(Ag + (size_t)ar1 * Ka + ac1, As + nb * A_TILE + ar1 * 40 + ac1);
            if (doB) async_b128(Wb + (size_t)(n0 + br) * Ka + kb + bc,
                                Bs + nb * B_TILE + br * 40 + bc);
            // drain stage s only (stage s+1 stays in flight)
            if (doB) S_WAIT_ASYNCCNT(0x3);
            else     S_WAIT_ASYNCCNT(0x2);
        } else {
            S_WAIT_ASYNCCNT(0x0);
        }
        __syncthreads();

        bffrag af, bf0, bf1;
        const bf16* ab  = As + buf * A_TILE + (w * 16 + r) * 40;
        const bf16* bb0 = Bs + buf * B_TILE + r * 40;
        const bf16* bb1 = Bs + buf * B_TILE + (16 + r) * 40;
        af.h[0]  = *(const v8bf*)(ab + koff);
        af.h[1]  = *(const v8bf*)(ab + koff + 16);
        bf0.h[0] = *(const v8bf*)(bb0 + koff);
        bf0.h[1] = *(const v8bf*)(bb0 + koff + 16);
        bf1.h[0] = *(const v8bf*)(bb1 + koff);
        bf1.h[1] = *(const v8bf*)(bb1 + koff + 16);

        acc0 = __builtin_amdgcn_wmma_f32_16x16x32_bf16(
                   false, af.v, false, bf0.v, (short)0, acc0, false, false);
        acc1 = __builtin_amdgcn_wmma_f32_16x16x32_bf16(
                   false, af.v, false, bf1.v, (short)0, acc1, false, false);
        __syncthreads();
    }

    // D layout: VGPR i -> M = i (lanes 0-15) / i+8 (lanes 16-31); N = lane&15
    const int colLocal = lane & 15;
    const int rowHi = (lane >> 4) * 8;
    const int rowBase = m0 + w * 16 + rowHi;
    const int o0 = n0 + colLocal;
    const int o1 = n0 + 16 + colLocal;
    const float sc0 = g[o0] * BN_SCALE_F, bz0 = bias[o0], bt0 = be[o0];
    const float sc1 = g[o1] * BN_SCALE_F, bz1 = bias[o1], bt1 = be[o1];
#pragma unroll
    for (int i = 0; i < 8; i++) {
        int row = rowBase + i;
        float h0 = (acc0[i] + bz0) * sc0 + bt0;
        float h1 = (acc1[i] + bz1) * sc1 + bt1;
        h0 = h0 > 0.f ? h0 : 0.f;
        h1 = h1 > 0.f ? h1 : 0.f;
        Out[(size_t)row * ldo + o0] = (bf16)h0;
        Out[(size_t)row * ldo + o1] = (bf16)h1;
    }
}

// ---------------------------------------------------------------------------
// Max pool over the K group samples; writes into a column slice of the
// concatenated per-level feature buffer.
// ---------------------------------------------------------------------------
__global__ void maxpool_kernel(const bf16* __restrict__ x, int ldx, int K, int O,
                               bf16* __restrict__ out, int ldo, int ooff, long total) {
    long t = (long)blockIdx.x * blockDim.x + threadIdx.x;
    if (t >= total) return;
    int o = (int)(t % O);
    long bs = t / O;
    float m = -3e38f;
    for (int k = 0; k < K; k++)
        m = fmaxf(m, (float)x[((size_t)bs * K + k) * ldx + o]);
    out[(size_t)bs * ldo + ooff + o] = (bf16)m;
}

// ---------------------------------------------------------------------------
// 3-NN search + inverse-distance weights for feature propagation.
// ---------------------------------------------------------------------------
__global__ void knn3_kernel(const float* __restrict__ x1, const float* __restrict__ x2,
                            int N1, int N2, int* __restrict__ idx3, float* __restrict__ w3,
                            int total) {
    int t = blockIdx.x * blockDim.x + threadIdx.x;
    if (t >= total) return;
    int b = t / N1;
    const float* q = x1 + (size_t)t * 3;
    const float* X = x2 + (size_t)b * N2 * 3;
    float qx = q[0], qy = q[1], qz = q[2];
    float d0 = 3e38f, d1 = 3e38f, d2 = 3e38f;
    int i0 = 0, i1 = 0, i2 = 0;
    for (int j = 0; j < N2; j++) {
        float dx = X[j * 3] - qx, dy = X[j * 3 + 1] - qy, dz = X[j * 3 + 2] - qz;
        float d = dx * dx + dy * dy + dz * dz;
        if (d < d0)      { d2 = d1; i2 = i1; d1 = d0; i1 = i0; d0 = d; i0 = j; }
        else if (d < d1) { d2 = d1; i2 = i1; d1 = d; i1 = j; }
        else if (d < d2) { d2 = d; i2 = j; }
    }
    float w0 = 1.f / (d0 + 1e-8f), w1 = 1.f / (d1 + 1e-8f), w2 = 1.f / (d2 + 1e-8f);
    float s = w0 + w1 + w2;
    idx3[t * 3] = i0; idx3[t * 3 + 1] = i1; idx3[t * 3 + 2] = i2;
    w3[t * 3] = w0 / s; w3[t * 3 + 1] = w1 / s; w3[t * 3 + 2] = w2 / s;
}

// ---------------------------------------------------------------------------
// FP concat: x = [points1 | interp(points2)] with zero K-padding.
// ---------------------------------------------------------------------------
__global__ void fp_concat_kernel(const bf16* __restrict__ p1, int ldp1, int C1,
                                 const bf16* __restrict__ p2, int ldp2, int C2,
                                 const int* __restrict__ idx3, const float* __restrict__ w3,
                                 int N1, int N2, bf16* __restrict__ x, int lda, long total) {
    long t = (long)blockIdx.x * blockDim.x + threadIdx.x;
    if (t >= total) return;
    int c = (int)(t % lda);
    long bn = t / lda;
    int b = (int)(bn / N1);
    float v = 0.f;
    if (c < C1) {
        v = (float)p1[(size_t)bn * ldp1 + c];
    } else if (c < C1 + C2) {
        int o = c - C1;
        const int* id = idx3 + bn * 3;
        const float* w = w3 + bn * 3;
        v = w[0] * (float)p2[((size_t)b * N2 + id[0]) * ldp2 + o]
          + w[1] * (float)p2[((size_t)b * N2 + id[1]) * ldp2 + o]
          + w[2] * (float)p2[((size_t)b * N2 + id[2]) * ldp2 + o];
    }
    x[t] = (bf16)v;
}

// ---------------------------------------------------------------------------
// Head: conv2 (128 -> 13) + log_softmax over classes, f32 output [B,N,13].
// ---------------------------------------------------------------------------
__global__ void head_kernel(const bf16* __restrict__ h, const float* __restrict__ W2,
                            const float* __restrict__ b2, float* __restrict__ out, int total) {
    int t = blockIdx.x * blockDim.x + threadIdx.x;
    if (t >= total) return;
    const bf16* row = h + (size_t)t * 128;
    float lg[NCLS];
    float m = -3e38f;
    for (int c = 0; c < NCLS; c++) {
        float a = b2[c];
        const float* w = W2 + (size_t)c * 128;
        for (int k = 0; k < 128; k++) a += (float)row[k] * w[k];
        lg[c] = a;
        m = fmaxf(m, a);
    }
    float s = 0.f;
    for (int c = 0; c < NCLS; c++) s += expf(lg[c] - m);
    float lse = logf(s);
    float* o = out + (size_t)t * NCLS;
    for (int c = 0; c < NCLS; c++) o[c] = lg[c] - m - lse;
}

// ---------------------------------------------------------------------------
// Host-side orchestration
// ---------------------------------------------------------------------------
struct Bump {
    char* base;
    size_t off;
    void* take(size_t n) {
        void* r = base + off;
        off = (off + n + 255) & ~(size_t)255;
        return r;
    }
};

struct MlpLayer { const float *W, *b, *g, *be; int O; };

static MlpLayer mk_layer(void* const* din, int base, int l, int O) {
    MlpLayer m;
    m.W  = (const float*)din[base + 4 * l + 0];
    m.b  = (const float*)din[base + 4 * l + 1];
    m.g  = (const float*)din[base + 4 * l + 2];
    m.be = (const float*)din[base + 4 * l + 3];
    m.O = O;
    return m;
}

static void run_gemm(hipStream_t st, Bump& ws, const bf16* A, long M, int C,
                     const MlpLayer& L, bf16* Out) {
    int Ka = align32i(C);
    bf16* Wb = (bf16*)ws.take((size_t)L.O * Ka * sizeof(bf16));
    long tw = (long)L.O * Ka;
    wconv_kernel<<<(unsigned)((tw + 255) / 256), 256, 0, st>>>(L.W, C, Wb, Ka, tw);
    dim3 grid((unsigned)(M / 128), (unsigned)(L.O / 32));
    wmma_gemm_bnrelu<<<grid, 256, 0, st>>>(A, Wb, L.b, L.g, L.be, Out, L.O, Ka);
}

static void sa_stage(hipStream_t st, Bump ws,
                     const float* xyzIn, const bf16* ptsIn, int Cp, int Nin, int S,
                     const float* radii, const int* Ks, const MlpLayer (*br)[3],
                     float* xyzOut, bf16* ptsOut, int ldOut) {
    int* fpsIdx = (int*)ws.take((size_t)BATCH * S * sizeof(int));
    fps_kernel<<<BATCH, 512, 0, st>>>(xyzIn, Nin, S, fpsIdx, xyzOut);

    int coloff = 0;
    for (int j = 0; j < 2; j++) {
        Bump bws = ws;                                  // per-branch scratch reset
        int K = Ks[j];
        float r2 = radii[j] * radii[j];
        int* gidx = (int*)bws.take((size_t)BATCH * S * K * sizeof(int));
        int totq = BATCH * S;
        ballquery_kernel<<<(totq + 255) / 256, 256, 0, st>>>(xyzIn, xyzOut, Nin, S, K, r2, gidx, totq);

        int Cin = Cp + 3, Ka0 = align32i(Cin);
        long M = (long)BATCH * S * K;
        long tg = M * Ka0;
        bf16* x0 = (bf16*)bws.take((size_t)tg * sizeof(bf16));
        group_kernel<<<(unsigned)((tg + 255) / 256), 256, 0, st>>>(
            xyzIn, xyzOut, ptsIn, Cp, Cp, gidx, Nin, S, K, x0, Ka0, tg);

        const bf16* cur = x0;
        int C = Cin;
        for (int l = 0; l < 3; l++) {
            bf16* o = (bf16*)bws.take((size_t)M * br[j][l].O * sizeof(bf16));
            run_gemm(st, bws, cur, M, C, br[j][l], o);
            cur = o;
            C = br[j][l].O;
        }
        long tp = (long)BATCH * S * C;
        maxpool_kernel<<<(unsigned)((tp + 255) / 256), 256, 0, st>>>(
            cur, C, K, C, ptsOut, ldOut, coloff, tp);
        coloff += C;
    }
}

static void fp_stage(hipStream_t st, Bump ws,
                     const float* x1, int N1, const float* x2, int N2,
                     const bf16* p1, int C1, const bf16* p2, int C2,
                     const MlpLayer* layers, int nl, bf16* out) {
    int* idx3 = (int*)ws.take((size_t)BATCH * N1 * 3 * sizeof(int));
    float* w3 = (float*)ws.take((size_t)BATCH * N1 * 3 * sizeof(float));
    int tq = BATCH * N1;
    knn3_kernel<<<(tq + 255) / 256, 256, 0, st>>>(x1, x2, N1, N2, idx3, w3, tq);

    int Cin = C1 + C2, Ka0 = align32i(Cin);
    long M = (long)BATCH * N1;
    long tc = M * Ka0;
    bf16* x = (bf16*)ws.take((size_t)tc * sizeof(bf16));
    fp_concat_kernel<<<(unsigned)((tc + 255) / 256), 256, 0, st>>>(
        p1, C1, C1, p2, C2, C2, idx3, w3, N1, N2, x, Ka0, tc);

    const bf16* cur = x;
    int C = Cin;
    for (int l = 0; l < nl; l++) {
        bf16* o = (l == nl - 1) ? out
                                : (bf16*)ws.take((size_t)M * layers[l].O * sizeof(bf16));
        run_gemm(st, ws, cur, M, C, layers[l], o);
        cur = o;
        C = layers[l].O;
    }
}

// jax pytree flattening of setup_inputs(): xyz, colors, then params with
// sorted dict keys: bn1, conv1, conv2, fp1..fp4, sa1..sa4 (tuples/lists in order).
enum {
    IN_XYZ = 0, IN_COLORS = 1,
    IN_BN1_G = 2, IN_BN1_B = 3,
    IN_CONV1_W = 4, IN_CONV1_B = 5,
    IN_CONV2_W = 6, IN_CONV2_B = 7,
    IN_FP1 = 8, IN_FP2 = 20, IN_FP3 = 28, IN_FP4 = 36,
    IN_SA1 = 44, IN_SA2 = 68, IN_SA3 = 92, IN_SA4 = 116
};

extern "C" void kernel_launch(void* const* d_in, const int* in_sizes, int n_in,
                              void* d_out, int out_size, void* d_ws, size_t ws_size,
                              hipStream_t stream) {
    (void)in_sizes; (void)n_in; (void)out_size; (void)ws_size;
    const float* xyz = (const float*)d_in[IN_XYZ];

    Bump ws{(char*)d_ws, 0};

    // persistent buffers
    float* xyzT0 = (float*)ws.take((size_t)BATCH * NPTS * 3 * sizeof(float));
    bf16*  pts0  = (bf16*)ws.take((size_t)BATCH * NPTS * 3 * sizeof(bf16));
    float* mstd  = (float*)ws.take(6 * 2 * sizeof(float));
    float* l1x = (float*)ws.take((size_t)BATCH * 4096 * 3 * sizeof(float));
    bf16*  l1p = (bf16*)ws.take((size_t)BATCH * 4096 * 192 * sizeof(bf16));
    float* l2x = (float*)ws.take((size_t)BATCH * 2048 * 3 * sizeof(float));
    bf16*  l2p = (bf16*)ws.take((size_t)BATCH * 2048 * 768 * sizeof(bf16));
    float* l3x = (float*)ws.take((size_t)BATCH * 512 * 3 * sizeof(float));
    bf16*  l3p = (bf16*)ws.take((size_t)BATCH * 512 * 1536 * sizeof(bf16));
    float* l4x = (float*)ws.take((size_t)BATCH * 128 * 3 * sizeof(float));
    bf16*  l4p = (bf16*)ws.take((size_t)BATCH * 128 * 3072 * sizeof(bf16));
    bf16*  l3pN = (bf16*)ws.take((size_t)BATCH * 512 * 512 * sizeof(bf16));
    bf16*  l2pN = (bf16*)ws.take((size_t)BATCH * 2048 * 256 * sizeof(bf16));
    bf16*  l1pN = (bf16*)ws.take((size_t)BATCH * 4096 * 128 * sizeof(bf16));
    bf16*  l0p  = (bf16*)ws.take((size_t)BATCH * NPTS * 128 * sizeof(bf16));
    bf16*  hbuf = (bf16*)ws.take((size_t)BATCH * NPTS * 128 * sizeof(bf16));
    Bump scratch = ws;   // everything below reuses this region per stage

    // ---- input normalization -> l0 ----
    meanstd_kernel<<<6, 256, 0, stream>>>(xyz, mstd);
    int tn = BATCH * NPTS * 3;
    norm_kernel<<<(tn + 255) / 256, 256, 0, stream>>>(xyz, mstd, xyzT0, pts0, tn);

    // ---- SA stages ----
    MlpLayer sa1[2][3] = {
        { mk_layer(d_in, IN_SA1 + 0, 0, 32), mk_layer(d_in, IN_SA1 + 0, 1, 32), mk_layer(d_in, IN_SA1 + 0, 2, 64) },
        { mk_layer(d_in, IN_SA1 + 12, 0, 64), mk_layer(d_in, IN_SA1 + 12, 1, 64), mk_layer(d_in, IN_SA1 + 12, 2, 128) } };
    MlpLayer sa2[2][3] = {
        { mk_layer(d_in, IN_SA2 + 0, 0, 128), mk_layer(d_in, IN_SA2 + 0, 1, 128), mk_layer(d_in, IN_SA2 + 0, 2, 256) },
        { mk_layer(d_in, IN_SA2 + 12, 0, 256), mk_layer(d_in, IN_SA2 + 12, 1, 256), mk_layer(d_in, IN_SA2 + 12, 2, 512) } };
    MlpLayer sa3[2][3] = {
        { mk_layer(d_in, IN_SA3 + 0, 0, 256), mk_layer(d_in, IN_SA3 + 0, 1, 256), mk_layer(d_in, IN_SA3 + 0, 2, 512) },
        { mk_layer(d_in, IN_SA3 + 12, 0, 512), mk_layer(d_in, IN_SA3 + 12, 1, 512), mk_layer(d_in, IN_SA3 + 12, 2, 1024) } };
    MlpLayer sa4[2][3] = {
        { mk_layer(d_in, IN_SA4 + 0, 0, 512), mk_layer(d_in, IN_SA4 + 0, 1, 512), mk_layer(d_in, IN_SA4 + 0, 2, 1024) },
        { mk_layer(d_in, IN_SA4 + 12, 0, 1024), mk_layer(d_in, IN_SA4 + 12, 1, 1024), mk_layer(d_in, IN_SA4 + 12, 2, 2048) } };

    const int Ks[2] = {32, 64};
    float r1[2] = {0.1f, 0.2f}, r2[2] = {0.2f, 0.4f}, r3[2] = {0.4f, 0.8f}, r4[2] = {0.8f, 1.6f};

    sa_stage(stream, scratch, xyzT0, pts0, 3,    NPTS, 4096, r1, Ks, sa1, l1x, l1p, 192);
    sa_stage(stream, scratch, l1x,   l1p,  192,  4096, 2048, r2, Ks, sa2, l2x, l2p, 768);
    sa_stage(stream, scratch, l2x,   l2p,  768,  2048, 512,  r3, Ks, sa3, l3x, l3p, 1536);
    sa_stage(stream, scratch, l3x,   l3p,  1536, 512,  128,  r4, Ks, sa4, l4x, l4p, 3072);

    // ---- FP stages ----
    MlpLayer fp4[2] = { mk_layer(d_in, IN_FP4, 0, 1024), mk_layer(d_in, IN_FP4, 1, 512) };
    MlpLayer fp3[2] = { mk_layer(d_in, IN_FP3, 0, 512),  mk_layer(d_in, IN_FP3, 1, 256) };
    MlpLayer fp2[2] = { mk_layer(d_in, IN_FP2, 0, 256),  mk_layer(d_in, IN_FP2, 1, 128) };
    MlpLayer fp1[3] = { mk_layer(d_in, IN_FP1, 0, 128),  mk_layer(d_in, IN_FP1, 1, 128),
                        mk_layer(d_in, IN_FP1, 2, 128) };

    fp_stage(stream, scratch, l3x, 512,  l4x, 128,  l3p, 1536, l4p, 3072, fp4, 2, l3pN);
    fp_stage(stream, scratch, l2x, 2048, l3x, 512,  l2p, 768,  l3pN, 512, fp3, 2, l2pN);
    fp_stage(stream, scratch, l1x, 4096, l2x, 2048, l1p, 192,  l2pN, 256, fp2, 2, l1pN);
    fp_stage(stream, scratch, xyzT0, NPTS, l1x, 4096, (const bf16*)nullptr, 0, l1pN, 128, fp1, 3, l0p);

    // ---- head: conv1 (+bn1+relu) via WMMA, then conv2 + log_softmax ----
    MlpLayer conv1L = { (const float*)d_in[IN_CONV1_W], (const float*)d_in[IN_CONV1_B],
                        (const float*)d_in[IN_BN1_G],   (const float*)d_in[IN_BN1_B], 128 };
    {
        Bump s2 = scratch;
        run_gemm(stream, s2, l0p, (long)BATCH * NPTS, 128, conv1L, hbuf);
    }
    int th = BATCH * NPTS;
    head_kernel<<<(th + 255) / 256, 256, 0, stream>>>(
        hbuf, (const float*)d_in[IN_CONV2_W], (const float*)d_in[IN_CONV2_B],
        (float*)d_out, th);
}
